// SemanticGuidedUpsampling_86431921865225
// MI455X (gfx1250) — matrix operand order repaired and checked by reading, not compiled, not run on hardware
//
#include <hip/hip_runtime.h>

// ---------------------------------------------------------------------------
// Types & helpers for CDNA5 WMMA (wave32, 16x16x32 bf16 -> f32)
// ---------------------------------------------------------------------------
typedef __attribute__((ext_vector_type(16))) __bf16         v16bf;
typedef __attribute__((ext_vector_type(8)))  float          v8f;
typedef __attribute__((ext_vector_type(16))) unsigned short v16us;

union BF16Vec {
  v16us u;
  v16bf b;
  uint4 q[2];
};

__device__ __forceinline__ unsigned short f2bf(float f) {
  union { float f; unsigned int u; } cv; cv.f = f;
  unsigned int x = cv.u;
  x += 0x7fffu + ((x >> 16) & 1u);   // round-to-nearest-even
  return (unsigned short)(x >> 16);
}

__device__ __forceinline__ unsigned int pack2bf(float lo, float hi) {
  return (unsigned int)f2bf(lo) | ((unsigned int)f2bf(hi) << 16);
}

__device__ __forceinline__ v8f vzero8() {
  v8f z = {0.f, 0.f, 0.f, 0.f, 0.f, 0.f, 0.f, 0.f};
  return z;
}

__device__ __forceinline__ v8f wmma_bf16(v16bf a, v16bf b, v8f c) {
  return __builtin_amdgcn_wmma_f32_16x16x32_bf16(false, a, false, b,
                                                 (short)0, c, false, false);
}

// Async global -> LDS copy of 16B for this lane (CDNA5 TDM-adjacent path,
// ISA §15.18.3 op 98, tracked by ASYNCcnt).  ldsOff is the wave-relative LDS
// byte address (dynamic-LDS kernels: segment starts at 0).
__device__ __forceinline__ void async_ld_lds_b128(unsigned int ldsOff,
                                                  const void* gaddr) {
  asm volatile("global_load_async_to_lds_b128 %0, %1, off"
               :
               : "v"(ldsOff), "v"(gaddr)
               : "memory");
}

__device__ __forceinline__ void wait_async0() {
  asm volatile("s_wait_asynccnt 0" ::: "memory");
}

// A tile (16x32 bf16) from a row-major bf16 buffer (16B-aligned rows).
// ISA 7.12.2: lanes 0-15 hold M=lane, K {0..7,16..23}; lanes 16-31 hold
// M=lane-16, K {8..15,24..31}.  Two contiguous 16B runs -> two b128 loads.
__device__ __forceinline__ v16bf load_a_bf16(const unsigned short* base, int ld) {
  const int lane = threadIdx.x & 31;
  const int row  = lane & 15;
  const int kb   = (lane >> 4) * 8;
  const unsigned short* p = base + row * ld + kb;
  BF16Vec r;
  r.q[0] = *(const uint4*)(p);
  r.q[1] = *(const uint4*)(p + 16);
  return r.b;
}

// B tile (32x16 bf16) from a pre-swizzled panel: for tile nt, lane l,
// the lane's 16 K-values are contiguous at ((nt*32 + l) * 16).
// -> two ds_read_b128, zero element shuffles.
__device__ __forceinline__ v16bf load_b_swz(const unsigned short* panel, int nt) {
  const int lane = threadIdx.x & 31;
  const unsigned short* p = panel + (((nt << 5) + lane) << 4);
  BF16Vec r;
  r.q[0] = *(const uint4*)(p);
  r.q[1] = *(const uint4*)(p + 8);
  return r.b;
}

__device__ __forceinline__ void store_swz_block(unsigned short* dst, const BF16Vec& r) {
  ((uint4*)dst)[0] = r.q[0];
  ((uint4*)dst)[1] = r.q[1];
}

// ---------------------------------------------------------------------------
// Problem constants (from setup_inputs)
// ---------------------------------------------------------------------------
#define BATCH  2
#define MPTS   4096
#define NPTS   8192
#define CIN    512
#define H1DIM  128
#define NCLS   20
#define H2DIM  256
#define OUTD   512
#define KDIM   544          // CIN + NCLS padded up to multiple of 32
#define BM     (BATCH * MPTS)
#define ROWS   (BATCH * NPTS)

// Pre-swizzled weight sizes (elements): (K/32) panels * (N/16) tiles * 512
#define PW1SEM_ELEMS  (16 * 8  * 512)   // w_sem1 512x128
#define PW2SEM_ELEMS  (4  * 2  * 512)   // w_sem2 128x20 -> 128x32
#define PUW1_ELEMS    (17 * 16 * 512)   // w_up1  532x256 -> 544x256
#define PUW2_ELEMS    (8  * 32 * 512)   // w_up2  256x512

// ---------------------------------------------------------------------------
// Prep kernel A: convert + swizzle a weight matrix into WMMA-B layout.
// ---------------------------------------------------------------------------
__global__ __launch_bounds__(256) void pack_weights_kernel(
    const float* __restrict__ W, unsigned short* __restrict__ dst,
    int ntl, int kmax, int ncols) {
  const int gid = blockIdx.x * 256 + threadIdx.x;
  const int p   = gid / (ntl * 32);
  const int rem = gid - p * (ntl * 32);
  const int nt  = rem >> 5, ln = rem & 31;
  const int col = nt * 16 + (ln & 15);
  const int kh  = (ln >> 4) * 8;
  BF16Vec r;
#pragma unroll
  for (int j = 0; j < 16; ++j) {
    int k = p * 32 + kh + (j < 8 ? j : j + 8);
    float v = (k < kmax && col < ncols) ? W[(size_t)k * ncols + col] : 0.f;
    r.u[j] = f2bf(v);
  }
  store_swz_block(dst + (size_t)gid * 16, r);
}

// ---------------------------------------------------------------------------
// Prep kernel B: fp32 -> bf16 streaming convert (8 elements / thread).
// ---------------------------------------------------------------------------
__global__ __launch_bounds__(256) void tobf16_kernel(
    const float* __restrict__ src, unsigned short* __restrict__ dst) {
  const size_t i0 = ((size_t)blockIdx.x * 256 + threadIdx.x) * 8;
  float4 f0 = *(const float4*)(src + i0);
  float4 f1 = *(const float4*)(src + i0 + 4);
  uint4 o;
  o.x = pack2bf(f0.x, f0.y);
  o.y = pack2bf(f0.z, f0.w);
  o.z = pack2bf(f1.x, f1.y);
  o.w = pack2bf(f1.z, f1.w);
  *(uint4*)(dst + i0) = o;
}

// ---------------------------------------------------------------------------
// Kernel 1: semantic head  logits = relu(X@W1+b1)@W2+b2
// WG = 256 thr = 8 waves; wave owns a 16-row slab; WG covers 128 rows.
// Dynamic LDS: [0,8K) W1 k-panel, [8K,16K) W2, [16K,48K) hidden staging.
// Weight panels arrive via global_load_async_to_lds_b128 (ASYNCcnt).
// ---------------------------------------------------------------------------
__global__ __launch_bounds__(256) void semantic_kernel(
    const unsigned short* __restrict__ Xbf, const unsigned short* __restrict__ pW1,
    const float* __restrict__ B1, const unsigned short* __restrict__ pW2,
    const float* __restrict__ B2, float* __restrict__ logits) {
  extern __shared__ __align__(16) unsigned short smem[];
  unsigned short* sW1p = smem;            // LDS byte offset 0,     8 KB
  unsigned short* sW2  = smem + 4096;     // LDS byte offset 8192,  8 KB
  unsigned short* sH   = smem + 8192;     // LDS byte offset 16384, 32 KB

  const int tid  = threadIdx.x;
  const int wave = tid >> 5;
  const int lane = tid & 31;
  const int row0 = blockIdx.x * 128;

  // One-time async copy of pre-swizzled W2 (8 KB); completion is covered by
  // the first k-step's s_wait_asynccnt + barrier.
  async_ld_lds_b128(8192u + tid * 16u,          pW2 + (size_t)tid * 8);
  async_ld_lds_b128(8192u + (tid + 256) * 16u,  pW2 + (size_t)(tid + 256) * 8);

  v8f acc[8];
#pragma unroll
  for (int i = 0; i < 8; ++i) acc[i] = vzero8();

  const unsigned short* arow = Xbf + (size_t)(row0 + wave * 16) * CIN;

  for (int k0 = 0; k0 < CIN; k0 += 32) {
    __syncthreads();                       // previous panel readers done
    {  // W1 k-panel: 8 KB async copy (512 x 16B blocks, 2 per thread)
      const unsigned short* pk = pW1 + (size_t)(k0 >> 5) * (8 * 32 * 16);
      async_ld_lds_b128(tid * 16u,         pk + (size_t)tid * 8);
      async_ld_lds_b128((tid + 256) * 16u, pk + (size_t)(tid + 256) * 8);
    }
    wait_async0();
    __syncthreads();                       // all waves' panel writes visible

    v16bf a = load_a_bf16(arow + k0, CIN);
#pragma unroll
    for (int nt = 0; nt < 8; ++nt)
      acc[nt] = wmma_bf16(a, load_b_swz(sW1p, nt), acc[nt]);
  }

  // bias + ReLU, stage hidden as bf16 (own-wave LDS region).
  unsigned short* hp = sH + wave * 16 * H1DIM;
#pragma unroll
  for (int nt = 0; nt < 8; ++nt) {
    int col    = nt * 16 + (lane & 15);
    float bias = B1[col];
#pragma unroll
    for (int r = 0; r < 8; ++r) {
      int row = r + (lane >> 4) * 8;
      hp[row * H1DIM + col] = f2bf(fmaxf(acc[nt][r] + bias, 0.f));
    }
  }
  __syncthreads();

  v8f acc2[2];
  acc2[0] = vzero8(); acc2[1] = vzero8();
#pragma unroll
  for (int kp = 0; kp < 4; ++kp) {
    v16bf a = load_a_bf16(hp + kp * 32, H1DIM);
#pragma unroll
    for (int nt = 0; nt < 2; ++nt)
      acc2[nt] = wmma_bf16(a, load_b_swz(sW2 + kp * 1024, nt), acc2[nt]);
  }

#pragma unroll
  for (int nt = 0; nt < 2; ++nt) {
    int n = nt * 16 + (lane & 15);
    if (n < NCLS) {
      float bias = B2[n];
#pragma unroll
      for (int r = 0; r < 8; ++r) {
        int row = row0 + wave * 16 + r + (lane >> 4) * 8;
        logits[(size_t)row * NCLS + n] = acc2[nt][r] + bias;
      }
    }
  }
}

// ---------------------------------------------------------------------------
// Kernel 2: kNN (top-3) + inverse-distance softmax weights + per-neighbor
// semantic softmax mean.  Source points cached in LDS (48 KB).
// ---------------------------------------------------------------------------
__global__ __launch_bounds__(256) void knn_kernel(
    const float* __restrict__ srcp, const float* __restrict__ tgtp,
    const float* __restrict__ logits, int* __restrict__ knn_idx,
    float* __restrict__ knn_w, float* __restrict__ semw) {
  __shared__ float sx[MPTS], sy[MPTS], sz[MPTS];

  const int b = blockIdx.y;
  const float* sp = srcp + (size_t)b * MPTS * 3;
  for (int i = threadIdx.x; i < MPTS; i += 256) {
    sx[i] = sp[i * 3 + 0];
    sy[i] = sp[i * 3 + 1];
    sz[i] = sp[i * 3 + 2];
  }
  __syncthreads();

  const int n = blockIdx.x * 256 + threadIdx.x;
  const int t = b * NPTS + n;
  const float tx = tgtp[(size_t)t * 3 + 0];
  const float ty = tgtp[(size_t)t * 3 + 1];
  const float tz = tgtp[(size_t)t * 3 + 2];

  float d0 = 1e30f, d1 = 1e30f, d2 = 1e30f;
  int   i0 = 0,     i1 = 0,     i2 = 0;
  for (int m = 0; m < MPTS; ++m) {
    float dx = tx - sx[m], dy = ty - sy[m], dz = tz - sz[m];
    float dd = dx * dx + dy * dy + dz * dz;
    if (dd < d0)      { d2 = d1; i2 = i1; d1 = d0; i1 = i0; d0 = dd; i0 = m; }
    else if (dd < d1) { d2 = d1; i2 = i1; d1 = dd; i1 = m; }
    else if (dd < d2) { d2 = dd; i2 = m; }
  }
  d0 = sqrtf(fmaxf(d0, 0.f));
  d1 = sqrtf(fmaxf(d1, 0.f));
  d2 = sqrtf(fmaxf(d2, 0.f));

  // softmax(-d) over the 3 neighbors (d0 is the min -> stable pivot)
  float e0 = 1.f, e1 = __expf(d0 - d1), e2 = __expf(d0 - d2);
  float inv = 1.f / (e0 + e1 + e2);
  knn_idx[t * 3 + 0] = i0;  knn_w[t * 3 + 0] = e0 * inv;
  knn_idx[t * 3 + 1] = i1;  knn_w[t * 3 + 1] = e1 * inv;
  knn_idx[t * 3 + 2] = i2;  knn_w[t * 3 + 2] = e2 * inv;

  // mean over k of softmax(logits[idx_k]) along the 20 classes
  float accs[NCLS];
#pragma unroll
  for (int c = 0; c < NCLS; ++c) accs[c] = 0.f;
  int idxs[3] = { i0, i1, i2 };
#pragma unroll
  for (int k = 0; k < 3; ++k) {
    const float* L = logits + ((size_t)b * MPTS + idxs[k]) * NCLS;
    float mx = L[0];
#pragma unroll
    for (int c = 1; c < NCLS; ++c) mx = fmaxf(mx, L[c]);
    float e[NCLS], s = 0.f;
#pragma unroll
    for (int c = 0; c < NCLS; ++c) { e[c] = __expf(L[c] - mx); s += e[c]; }
    float is = 1.f / s;
#pragma unroll
    for (int c = 0; c < NCLS; ++c) accs[c] += e[c] * is;
  }
  const float third = (1.f / 3.f);
#pragma unroll
  for (int c = 0; c < NCLS; ++c) semw[(size_t)t * NCLS + c] = accs[c] * third;
}

// ---------------------------------------------------------------------------
// Kernel 3: build bf16 combined rows [ROWS x KDIM]: weighted feature fusion
// (cols 0..511) + semantic weights (512..531) + zero pad (532..543).
// One wave per target point; float2 reads, packed b32 stores.
// ---------------------------------------------------------------------------
__global__ __launch_bounds__(256) void fuse_kernel(
    const float* __restrict__ feat, const int* __restrict__ knn_idx,
    const float* __restrict__ knn_w, const float* __restrict__ semw,
    unsigned short* __restrict__ combined) {
  const int t    = blockIdx.x * 8 + (threadIdx.x >> 5);
  const int lane = threadIdx.x & 31;
  const int b    = t >> 13;   // NPTS == 8192 targets per batch

  const int   i0 = knn_idx[t * 3 + 0];
  const int   i1 = knn_idx[t * 3 + 1];
  const int   i2 = knn_idx[t * 3 + 2];
  const float w0 = knn_w[t * 3 + 0];
  const float w1 = knn_w[t * 3 + 1];
  const float w2 = knn_w[t * 3 + 2];

  const float2* F0 = (const float2*)(feat + ((size_t)b * MPTS + i0) * CIN);
  const float2* F1 = (const float2*)(feat + ((size_t)b * MPTS + i1) * CIN);
  const float2* F2 = (const float2*)(feat + ((size_t)b * MPTS + i2) * CIN);
  unsigned int* out32 = (unsigned int*)(combined + (size_t)t * KDIM);

#pragma unroll
  for (int c2 = lane; c2 < CIN / 2; c2 += 32) {
    float2 a0 = F0[c2], a1 = F1[c2], a2 = F2[c2];
    float lo = w0 * a0.x + w1 * a1.x + w2 * a2.x;
    float hi = w0 * a0.y + w1 * a1.y + w2 * a2.y;
    out32[c2] = pack2bf(lo, hi);
  }
  if (lane < 16) {   // cols 512..543: semantic weights + zero pad
    int c = lane * 2;
    float lo = (c     < NCLS) ? semw[(size_t)t * NCLS + c]     : 0.f;
    float hi = (c + 1 < NCLS) ? semw[(size_t)t * NCLS + c + 1] : 0.f;
    out32[CIN / 2 + lane] = pack2bf(lo, hi);
  }
}

// ---------------------------------------------------------------------------
// Kernel 4: upsample MLP  out = relu(combined@W1+b1)@W2+b2
// WG = 256 thr = 8 waves, 128 rows per WG.  Dynamic LDS: [0,16K) weight panel,
// [16K,80K) hidden staging.  Panels arrive via async global->LDS copies.
// ---------------------------------------------------------------------------
__global__ __launch_bounds__(256) void upsample_kernel(
    const unsigned short* __restrict__ combined,
    const unsigned short* __restrict__ pW1, const float* __restrict__ B1,
    const unsigned short* __restrict__ pW2, const float* __restrict__ B2,
    float* __restrict__ out) {
  extern __shared__ __align__(16) unsigned short smem[];
  unsigned short* sWp = smem;               // LDS byte offset 0, 16 KB
  unsigned short* sH  = smem + 32 * H2DIM;  // LDS byte offset 16384, 64 KB

  const int tid  = threadIdx.x;
  const int wave = tid >> 5;
  const int lane = tid & 31;
  const int row0 = blockIdx.x * 128 + wave * 16;

  v8f acc[16];
#pragma unroll
  for (int i = 0; i < 16; ++i) acc[i] = vzero8();

  const unsigned short* arow = combined + (size_t)row0 * KDIM;

  // Phase 1: hidden = relu(combined @ W1 + b1), K = 544, N = 256
  for (int k0 = 0; k0 < KDIM; k0 += 32) {
    __syncthreads();
    {  // 16 KB panel: 1024 x 16B async copies, 4 per thread
      const unsigned short* pk = pW1 + (size_t)(k0 >> 5) * (16 * 32 * 16);
#pragma unroll
      for (int i = 0; i < 4; ++i)
        async_ld_lds_b128((tid + i * 256) * 16u, pk + (size_t)(tid + i * 256) * 8);
    }
    wait_async0();
    __syncthreads();

    v16bf a = load_a_bf16(arow + k0, KDIM);
#pragma unroll
    for (int nt = 0; nt < 16; ++nt)
      acc[nt] = wmma_bf16(a, load_b_swz(sWp, nt), acc[nt]);
  }

  // bias + ReLU, stage hidden per-wave for A-layout reload.
  unsigned short* hp = sH + wave * 16 * H2DIM;
#pragma unroll
  for (int nt = 0; nt < 16; ++nt) {
    int col    = nt * 16 + (lane & 15);
    float bias = B1[col];
#pragma unroll
    for (int r = 0; r < 8; ++r) {
      int row = r + (lane >> 4) * 8;
      hp[row * H2DIM + col] = f2bf(fmaxf(acc[nt][r] + bias, 0.f));
    }
  }

  // Phase 2: out = hidden @ W2 + b2, K = 256, N = 512 in two halves of 256.
  for (int h = 0; h < 2; ++h) {
    v8f acc2[16];
#pragma unroll
    for (int i = 0; i < 16; ++i) acc2[i] = vzero8();

    for (int k0 = 0; k0 < H2DIM; k0 += 32) {
      __syncthreads();
      {  // 16 tiles of this half: contiguous 16 KB, async copy
        const unsigned short* pk = pW2 + ((size_t)(k0 >> 5) * 32 + h * 16) * 512;
#pragma unroll
        for (int i = 0; i < 4; ++i)
          async_ld_lds_b128((tid + i * 256) * 16u, pk + (size_t)(tid + i * 256) * 8);
      }
      wait_async0();
      __syncthreads();

      v16bf a = load_a_bf16(hp + k0, H2DIM);
#pragma unroll
      for (int nt = 0; nt < 16; ++nt)
        acc2[nt] = wmma_bf16(a, load_b_swz(sWp, nt), acc2[nt]);
    }

#pragma unroll
    for (int nt = 0; nt < 16; ++nt) {
      int n      = h * H2DIM + nt * 16 + (lane & 15);
      float bias = B2[n];
#pragma unroll
      for (int r = 0; r < 8; ++r) {
        int row = row0 + r + (lane >> 4) * 8;
        out[(size_t)row * OUTD + n] = acc2[nt][r] + bias;
      }
    }
  }
}

// ---------------------------------------------------------------------------
// Launcher
// ---------------------------------------------------------------------------
extern "C" void kernel_launch(void* const* d_in, const int* in_sizes, int n_in,
                              void* d_out, int out_size, void* d_ws, size_t ws_size,
                              hipStream_t stream) {
  (void)in_sizes; (void)n_in; (void)out_size; (void)ws_size;

  const float* src_points   = (const float*)d_in[0];
  const float* tgt_points   = (const float*)d_in[1];
  const float* src_features = (const float*)d_in[2];
  const float* w_sem1 = (const float*)d_in[3];
  const float* b_sem1 = (const float*)d_in[4];
  const float* w_sem2 = (const float*)d_in[5];
  const float* b_sem2 = (const float*)d_in[6];
  const float* w_up1  = (const float*)d_in[7];
  const float* b_up1  = (const float*)d_in[8];
  const float* w_up2  = (const float*)d_in[9];
  const float* b_up2  = (const float*)d_in[10];

  float* upsampled = (float*)d_out;                               // [2,8192,512]
  float* logits    = upsampled + (size_t)ROWS * OUTD;             // [2,4096,20]

  // Workspace partition (~28.6 MB)
  char* ws = (char*)d_ws;
  int*   knn_idx = (int*)ws;            ws += (size_t)ROWS * 3 * sizeof(int);
  float* knn_w   = (float*)ws;          ws += (size_t)ROWS * 3 * sizeof(float);
  float* semw    = (float*)ws;          ws += (size_t)ROWS * NCLS * sizeof(float);
  unsigned short* combined = (unsigned short*)ws;  // [ROWS, KDIM] bf16
                                        ws += (size_t)ROWS * KDIM * sizeof(unsigned short);
  unsigned short* Xbf    = (unsigned short*)ws;  ws += (size_t)BM * CIN * sizeof(unsigned short);
  unsigned short* pW1sem = (unsigned short*)ws;  ws += (size_t)PW1SEM_ELEMS * sizeof(unsigned short);
  unsigned short* pW2sem = (unsigned short*)ws;  ws += (size_t)PW2SEM_ELEMS * sizeof(unsigned short);
  unsigned short* pUW1   = (unsigned short*)ws;  ws += (size_t)PUW1_ELEMS * sizeof(unsigned short);
  unsigned short* pUW2   = (unsigned short*)ws;

  // --- One-time format prep (weights -> swizzled bf16, X -> bf16) ---
  pack_weights_kernel<<<PW1SEM_ELEMS / 16 / 256, 256, 0, stream>>>(
      w_sem1, pW1sem, H1DIM / 16, CIN, H1DIM);            // 512x128
  pack_weights_kernel<<<PW2SEM_ELEMS / 16 / 256, 256, 0, stream>>>(
      w_sem2, pW2sem, 32 / 16, H1DIM, NCLS);              // 128x20 -> 128x32
  pack_weights_kernel<<<PUW1_ELEMS / 16 / 256, 256, 0, stream>>>(
      w_up1, pUW1, H2DIM / 16, CIN + NCLS, H2DIM);        // 532x256 -> 544x256
  pack_weights_kernel<<<PUW2_ELEMS / 16 / 256, 256, 0, stream>>>(
      w_up2, pUW2, OUTD / 16, H2DIM, OUTD);               // 256x512
  tobf16_kernel<<<(BM * CIN / 8) / 256, 256, 0, stream>>>(src_features, Xbf);

  // --- Main pipeline ---
  const size_t smem1 = (size_t)(8 * 32 * 16 + 4 * 2 * 32 * 16 + 8 * 16 * H1DIM) *
                       sizeof(unsigned short);   // 48 KB
  semantic_kernel<<<BM / 128, 256, smem1, stream>>>(Xbf, pW1sem, b_sem1,
                                                    pW2sem, b_sem2, logits);
  knn_kernel<<<dim3(NPTS / 256, BATCH), 256, 0, stream>>>(src_points, tgt_points,
                                                          logits, knn_idx, knn_w, semw);
  fuse_kernel<<<ROWS / 8, 256, 0, stream>>>(src_features, knn_idx, knn_w, semw, combined);

  const size_t smem4 = (size_t)(32 * H2DIM + 8 * 16 * H2DIM) * sizeof(unsigned short);
  upsample_kernel<<<ROWS / 128, 256, smem4, stream>>>(combined, pUW1, b_up1,
                                                      pUW2, b_up2, upsampled);
}
